// MS3M_13082470383824
// MI455X (gfx1250) — matrix operand order repaired and checked by reading, not compiled, not run on hardware
//
#include <hip/hip_runtime.h>
#include <hip/hip_bf16.h>

// Problem constants (match reference)
#define BB 4
#define TT 4096
#define DD 256
#define EE 12
#define NTOK (BB * TT)          // 16384 tokens
#define KBIG (EE * DD)          // 3072 reduction dim for big GEMM

typedef __bf16 bf16_t;
typedef __attribute__((ext_vector_type(16))) __bf16 v16bf;
typedef __attribute__((ext_vector_type(8)))  __bf16 v8bf;
typedef __attribute__((ext_vector_type(8)))  float  v8f;
typedef unsigned int v4u __attribute__((ext_vector_type(4)));
typedef int          v8i __attribute__((ext_vector_type(8)));
typedef int          v4i __attribute__((ext_vector_type(4)));

// ---------------------------------------------------------------------------
// TDM: async 2D tile load (16 rows x 256 bf16) from row-major [*, 3072] tensor
// into LDS, per CDNA5 ISA D# layout (CDNA5 ISA).
// Issued by one wave; completion via TENSORcnt.
// ---------------------------------------------------------------------------
__device__ __forceinline__ void tdm_load_tile16x256(const bf16_t* gsrc,
                                                    unsigned lds_byte_addr) {
    unsigned long long ga = (unsigned long long)(uintptr_t)gsrc;
    v4u g0;
    g0[0] = 1u;                                   // count=1, user mode, no gather
    g0[1] = lds_byte_addr;                        // lds_addr [63:32]
    g0[2] = (unsigned)(ga & 0xFFFFFFFFu);         // global_addr[31:0]
    g0[3] = (unsigned)((ga >> 32) & 0x1FFFFFFu)   // global_addr[56:32]
          | (2u << 30);                           // type=2 ("image")
    v8i g1;
    g1[0] = 0x00010000;                           // data_size=1 (2 bytes), mask=0
    g1[1] = (int)((KBIG & 0xFFFF) << 16);         // tensor_dim0[15:0] @ bits 63:48
    g1[2] = (int)(((KBIG >> 16) & 0xFFFF)         // tensor_dim0[31:16]
          | ((NTOK & 0xFFFF) << 16));             // tensor_dim1[15:0]
    g1[3] = (int)(((NTOK >> 16) & 0xFFFF)         // tensor_dim1[31:16]
          | (256u << 16));                        // tile_dim0 = 256
    g1[4] = 16;                                   // tile_dim1 = 16, tile_dim2 = 0
    g1[5] = KBIG;                                 // tensor_dim0_stride[31:0]
    g1[6] = 0;                                    // stride0[47:32]=0, stride1 lo=0
    g1[7] = 0;
    v4i gz = {0, 0, 0, 0};
#if __clang_major__ >= 23
    v8i gz8 = {0, 0, 0, 0, 0, 0, 0, 0};
    __builtin_amdgcn_tensor_load_to_lds(g0, g1, gz, gz, gz8, 0);
#else
    __builtin_amdgcn_tensor_load_to_lds(g0, g1, gz, gz, 0);
#endif
}

// ---------------------------------------------------------------------------
// Kernel 0a: pack proj_W [E,D,D] fp32 -> WbigT bf16 [256 (=f)][3072 (=e*256+d)]
// ---------------------------------------------------------------------------
__global__ __launch_bounds__(256)
void pack_projW(const float* __restrict__ projW, bf16_t* __restrict__ WbigT) {
    int idx = blockIdx.x * 256 + threadIdx.x;      // over 256*3072
    int f = idx / KBIG;
    int k = idx % KBIG;                            // k = e*256 + d
    WbigT[idx] = (bf16_t)projW[(size_t)k * DD + f];
}

// Kernel 0b: pack out_W [D,D] fp32 -> outWT bf16 [256 (=f)][256 (=d)]
__global__ __launch_bounds__(256)
void pack_outW(const float* __restrict__ outW, bf16_t* __restrict__ outWT) {
    int idx = blockIdx.x * 256 + threadIdx.x;      // over 256*256
    int f = idx / DD;
    int d = idx % DD;
    outWT[idx] = (bf16_t)outW[(size_t)d * DD + f];
}

// ---------------------------------------------------------------------------
// Kernel 1: router softmax + causal depthwise convs + router scaling.
//   A[bt][e*256+d]  = bf16( w_e * conv_e(x)[bt,d] )
//   biasTok[bt][f]  = sum_e w_e * proj_b[e][f]
// ---------------------------------------------------------------------------
__global__ __launch_bounds__(256)
void router_conv_scale(const float* __restrict__ x,
                       const float* __restrict__ c5,
                       const float* __restrict__ c9,
                       const float* __restrict__ c17,
                       const float* __restrict__ rW,
                       const float* __restrict__ rB,
                       const float* __restrict__ projB,
                       bf16_t* __restrict__ A,
                       float* __restrict__ biasTok)
{
    const int bt = blockIdx.x;
    const int b  = bt / TT;
    const int t  = bt % TT;
    const int d  = threadIdx.x;

    __shared__ float s_logit[EE];
    __shared__ float s_w[EE];
    if (d < EE) s_logit[d] = rB[d];
    __syncthreads();

    const float xv = x[(size_t)bt * DD + d];
    #pragma unroll
    for (int e = 0; e < EE; ++e)
        atomicAdd(&s_logit[e], xv * rW[d * EE + e]);
    __syncthreads();

    if (d == 0) {
        float m = s_logit[0];
        #pragma unroll
        for (int e = 1; e < EE; ++e) m = fmaxf(m, s_logit[e]);
        float s = 0.f;
        #pragma unroll
        for (int e = 0; e < EE; ++e) { float ev = __expf(s_logit[e] - m); s_w[e] = ev; s += ev; }
        float inv = 1.f / s;
        #pragma unroll
        for (int e = 0; e < EE; ++e) s_w[e] *= inv;
    }
    __syncthreads();

    float win[17];
    #pragma unroll
    for (int j = 0; j < 17; ++j) {
        int tt = t - 16 + j;
        win[j] = (tt >= 0) ? x[((size_t)b * TT + tt) * DD + d] : 0.f;
    }

    bf16_t* arow = A + (size_t)bt * KBIG;
    #pragma unroll
    for (int e = 0; e < 4; ++e) {   // k = 5
        float s = 0.f;
        #pragma unroll
        for (int j = 0; j < 5; ++j) s += win[12 + j] * c5[(e * 5 + j) * DD + d];
        arow[e * DD + d] = (bf16_t)(s_w[e] * s);
    }
    #pragma unroll
    for (int e = 0; e < 4; ++e) {   // k = 9
        float s = 0.f;
        #pragma unroll
        for (int j = 0; j < 9; ++j) s += win[8 + j] * c9[(e * 9 + j) * DD + d];
        arow[(4 + e) * DD + d] = (bf16_t)(s_w[4 + e] * s);
    }
    #pragma unroll
    for (int e = 0; e < 4; ++e) {   // k = 17
        float s = 0.f;
        #pragma unroll
        for (int j = 0; j < 17; ++j) s += win[j] * c17[(e * 17 + j) * DD + d];
        arow[(8 + e) * DD + d] = (bf16_t)(s_w[8 + e] * s);
    }

    float bsum = 0.f;
    #pragma unroll
    for (int e = 0; e < EE; ++e) bsum += s_w[e] * projB[e * DD + d];
    biasTok[(size_t)bt * DD + d] = bsum;
}

// ---------------------------------------------------------------------------
// Fused kernel: big GEMM (K=3072, A tiles DMA'd into LDS by the TDM, double
// buffered) -> +biasTok -> LayerNorm (block-wide LDS reduction) -> final GEMM
// (K=256, A from LDS) -> +out_b.
//
// Block = 16 tokens x 256 features, 8 waves, each wave owns 2 WMMA n-tiles.
// ---------------------------------------------------------------------------
__global__ __launch_bounds__(256)
void fused_moe_ln_out(const bf16_t* __restrict__ A,       // [NTOK][3072]
                      const bf16_t* __restrict__ WbigT,   // [256][3072]
                      const bf16_t* __restrict__ outWT,   // [256][256]
                      const float*  __restrict__ biasTok, // [NTOK][256]
                      const float*  __restrict__ gamma,   // [256]
                      const float*  __restrict__ beta,    // [256]
                      const float*  __restrict__ outB,    // [256]
                      float* __restrict__ Cout)           // [NTOK][256]
{
    __shared__ bf16_t As[2][16][256];   // double-buffered A tile (16 KB)
    __shared__ float  Ms[16][256];      // mixed tile             (16 KB)
    __shared__ bf16_t Ns[16][256];      // normed tile            (8 KB)
    __shared__ float  red[16][16][2];   // per-row partial sums   (2 KB)
    __shared__ float  stat[16][2];      // (mu, inv_sigma) per row

    const int tid   = threadIdx.x;
    const int wave  = tid >> 5;
    const int lane  = tid & 31;
    const int r16   = lane & 15;
    const int hi    = lane >> 4;
    const int mbase = blockIdx.x * 16;
    const int nbase = wave * 32;

    const unsigned asbase = (unsigned)(uintptr_t)(&As[0][0][0]);
    const bf16_t* Atile = A + (size_t)mbase * KBIG;

    // ---- Stage 1: mixed = A @ WbigT^T  (K = 3072), TDM-staged A tiles ----
    v8f acc0 = {};
    v8f acc1 = {};
    const bf16_t* Bcol0 = WbigT + (size_t)(nbase + r16) * KBIG;
    const bf16_t* Bcol1 = WbigT + (size_t)(nbase + 16 + r16) * KBIG;

    if (wave == 0)
        tdm_load_tile16x256(Atile, asbase);           // prefetch chunk 0

    for (int c = 0; c < KBIG / 256; ++c) {
        const int cur = c & 1;
        if (wave == 0)
            __builtin_amdgcn_s_wait_tensorcnt(0);     // chunk c landed in LDS
        __syncthreads();                              // visible to all waves
        if (wave == 0 && c + 1 < KBIG / 256)          // overlap next DMA
            tdm_load_tile16x256(Atile + (c + 1) * 256,
                                asbase + (unsigned)(1 - cur) * (16 * 256 * 2));

        const int kk0 = c * 256;
        #pragma unroll
        for (int ks = 0; ks < 256; ks += 32) {
            v8bf alo = *(const v8bf*)&As[cur][r16][ks + 16 * hi];
            v8bf ahi = *(const v8bf*)&As[cur][r16][ks + 16 * hi + 8];
            v16bf a;
            #pragma unroll
            for (int i = 0; i < 8; ++i) { a[i] = alo[i]; a[8 + i] = ahi[i]; }

            v16bf b0 = *(const v16bf*)(Bcol0 + kk0 + ks + 16 * hi);
            v16bf b1 = *(const v16bf*)(Bcol1 + kk0 + ks + 16 * hi);

            acc0 = __builtin_amdgcn_wmma_f32_16x16x32_bf16(
                       false, a, false, b0, (short)0, acc0, false, false);
            acc1 = __builtin_amdgcn_wmma_f32_16x16x32_bf16(
                       false, a, false, b1, (short)0, acc1, false, false);
        }
        __syncthreads();                              // done reading buf `cur`
    }

    // ---- Stage 1 epilogue: + biasTok, spill mixed tile to LDS ----
    #pragma unroll
    for (int r = 0; r < 8; ++r) {
        const int m  = hi * 8 + r;
        const int gm = mbase + m;
        const int n0 = nbase + r16;
        const int n1 = nbase + 16 + r16;
        Ms[m][n0] = acc0[r] + biasTok[(size_t)gm * DD + n0];
        Ms[m][n1] = acc1[r] + biasTok[(size_t)gm * DD + n1];
    }
    __syncthreads();

    // ---- Stage 2: LayerNorm across D=256 per row (block-wide) ----
    const int lnrow = tid >> 4;         // 0..15
    const int lnc   = tid & 15;         // 0..15
    {
        float s = 0.f, q = 0.f;
        #pragma unroll
        for (int j = 0; j < 16; ++j) {
            float v = Ms[lnrow][lnc + 16 * j];
            s += v;
            q += v * v;
        }
        red[lnrow][lnc][0] = s;
        red[lnrow][lnc][1] = q;
    }
    __syncthreads();
    if (lnc == 0) {
        float S = 0.f, Q = 0.f;
        #pragma unroll
        for (int j = 0; j < 16; ++j) { S += red[lnrow][j][0]; Q += red[lnrow][j][1]; }
        const float mu  = S * (1.f / DD);
        const float var = Q * (1.f / DD) - mu * mu;
        stat[lnrow][0] = mu;
        stat[lnrow][1] = rsqrtf(var + 1e-5f);
    }
    __syncthreads();
    {
        const float mu = stat[lnrow][0];
        const float is = stat[lnrow][1];
        #pragma unroll
        for (int j = 0; j < 16; ++j) {
            const int col = lnc + 16 * j;
            float v = Ms[lnrow][col];
            Ns[lnrow][col] = (bf16_t)((v - mu) * is * gamma[col] + beta[col]);
        }
    }
    __syncthreads();

    // ---- Stage 3: out = normed @ outWT^T + out_b  (K = 256, A from LDS) ----
    v8f o0 = {};
    v8f o1 = {};
    const bf16_t* Ocol0 = outWT + (size_t)(nbase + r16) * DD;
    const bf16_t* Ocol1 = outWT + (size_t)(nbase + 16 + r16) * DD;

    #pragma unroll
    for (int ks = 0; ks < DD; ks += 32) {
        v8bf alo = *(const v8bf*)&Ns[r16][ks + 16 * hi];
        v8bf ahi = *(const v8bf*)&Ns[r16][ks + 16 * hi + 8];
        v16bf a;
        #pragma unroll
        for (int i = 0; i < 8; ++i) { a[i] = alo[i]; a[8 + i] = ahi[i]; }

        v16bf b0 = *(const v16bf*)(Ocol0 + ks + 16 * hi);
        v16bf b1 = *(const v16bf*)(Ocol1 + ks + 16 * hi);

        o0 = __builtin_amdgcn_wmma_f32_16x16x32_bf16(
                 false, a, false, b0, (short)0, o0, false, false);
        o1 = __builtin_amdgcn_wmma_f32_16x16x32_bf16(
                 false, a, false, b1, (short)0, o1, false, false);
    }

    #pragma unroll
    for (int r = 0; r < 8; ++r) {
        const int gm = mbase + hi * 8 + r;
        const int n0 = nbase + r16;
        const int n1 = nbase + 16 + r16;
        Cout[(size_t)gm * DD + n0] = o0[r] + outB[n0];
        Cout[(size_t)gm * DD + n1] = o1[r] + outB[n1];
    }
}

// ---------------------------------------------------------------------------
// Launcher
// ---------------------------------------------------------------------------
extern "C" void kernel_launch(void* const* d_in, const int* in_sizes, int n_in,
                              void* d_out, int out_size, void* d_ws, size_t ws_size,
                              hipStream_t stream) {
    const float* x        = (const float*)d_in[0];
    const float* conv_k5  = (const float*)d_in[1];
    const float* conv_k9  = (const float*)d_in[2];
    const float* conv_k17 = (const float*)d_in[3];
    const float* proj_W   = (const float*)d_in[4];
    const float* proj_b   = (const float*)d_in[5];
    const float* router_W = (const float*)d_in[6];
    const float* router_b = (const float*)d_in[7];
    const float* out_W    = (const float*)d_in[8];
    const float* out_b    = (const float*)d_in[9];
    const float* gamma    = (const float*)d_in[10];
    const float* beta     = (const float*)d_in[11];
    float* out = (float*)d_out;

    // Workspace layout (bytes)
    char* ws = (char*)d_ws;
    size_t off = 0;
    bf16_t* A       = (bf16_t*)(ws + off); off += (size_t)NTOK * KBIG * sizeof(bf16_t); // 96 MB
    bf16_t* WbigT   = (bf16_t*)(ws + off); off += (size_t)DD * KBIG * sizeof(bf16_t);   // 1.5 MB
    bf16_t* outWT   = (bf16_t*)(ws + off); off += (size_t)DD * DD * sizeof(bf16_t);     // 128 KB
    float*  biasTok = (float*) (ws + off); off += (size_t)NTOK * DD * sizeof(float);    // 16 MB
    (void)ws_size; (void)in_sizes; (void)n_in; (void)out_size;

    // 0) pack weights to bf16, transposed to [f][k]
    pack_projW<<<(DD * KBIG) / 256, 256, 0, stream>>>(proj_W, WbigT);
    pack_outW <<<(DD * DD) / 256,   256, 0, stream>>>(out_W, outWT);

    // 1) router softmax + convs + scaling -> A, biasTok
    router_conv_scale<<<NTOK, 256, 0, stream>>>(x, conv_k5, conv_k9, conv_k17,
                                                router_W, router_b, proj_b,
                                                A, biasTok);

    // 2) fused: big GEMM (TDM-staged A) + bias + LayerNorm + final GEMM
    fused_moe_ln_out<<<NTOK / 16, 256, 0, stream>>>(A, WbigT, outWT, biasTok,
                                                    gamma, beta, out_b, out);
}
